// CRFLayer_74191265071870
// MI455X (gfx1250) — matrix operand order repaired and checked by reading, not compiled, not run on hardware
//
#include <hip/hip_runtime.h>
#include <hip/hip_bf16.h>

typedef __attribute__((ext_vector_type(16))) _Float16 v16h;
typedef __attribute__((ext_vector_type(8)))  float    v8f;
typedef __attribute__((ext_vector_type(8)))  unsigned v8u;

#define NL    48
#define SEQ   1024
#define BATCH 512

// ---- pack two f32 -> one dword of 2xf16 ----
__device__ __forceinline__ unsigned pack2h(float a, float b) {
  unsigned short ua = __builtin_bit_cast(unsigned short, (_Float16)a);
  unsigned short ub = __builtin_bit_cast(unsigned short, (_Float16)b);
  return (unsigned)ua | ((unsigned)ub << 16);
}

// ---- lanes 0-15 <-> 16-31 exchange via v_permlanex16_b32 (VALU, no DS) ----
__device__ __forceinline__ unsigned permx16_u32(unsigned v) {
  return (unsigned)__builtin_amdgcn_permlanex16(
      (int)v, (int)v, 0x76543210, 0xFEDCBA98, false, false);
}
__device__ __forceinline__ float permx16_f32(float v) {
  return __builtin_bit_cast(float,
      permx16_u32(__builtin_bit_cast(unsigned, v)));
}

// ============================================================================
// Forward algorithm. One wave (32 threads) per 16-batch tile.
//
// State in WMMA C/D layout: alpha[j, b], j = 16*jt + 8*hi + r, b = b0+lo.
// Scaled-linear recurrence:
//   w     = (E^T @ alpha) .* exp(emit_t)      (E = exp(T), K padded 48->64)
//   mx[b] = max_j w;  alpha' = w/mx;  o[b] += log(mx)    (masked)
// A-operand (E^T) is loop-invariant; B rebuilt from alpha each step with
// 12 cvt_pk + 12 permlanex16 + 16 cndmask (all VALU, nothing on DS).
// ============================================================================
__global__ __launch_bounds__(32) void crf_forward_kernel(
    const float* __restrict__ emis, const int* __restrict__ mask,
    const float* __restrict__ trans, const float* __restrict__ startT,
    const float* __restrict__ endT, float* __restrict__ fwd_out)
{
  const int lane = threadIdx.x;
  const int lo   = lane & 15;
  const int hi   = lane >> 4;
  const int b0   = blockIdx.x * 16;
  const int bb   = b0 + lo;                 // this lane's batch column

  // ---- A = E^T in WMMA 16-bit A layout: A[m=j, k=i], K padded 48->64 ----
  // This code's A order (B matches it): M = lane&15, K = 16*hi + h within a 32-K chunk.
  v16h A[3][2];
#pragma unroll
  for (int jt = 0; jt < 3; ++jt)
#pragma unroll
    for (int kc = 0; kc < 2; ++kc)
#pragma unroll
      for (int h = 0; h < 16; ++h) {
        int i = 32 * kc + 16 * hi + h;   // K index (label i)
        int j = 16 * jt + lo;                             // M index (label j)
        float v = (i < NL) ? __expf(trans[i * NL + j]) : 0.0f;
        A[jt][kc][h] = (_Float16)v;
      }

  // ---- t = 0 init: score0 = startT[j] + emis[b,0,j] ----
  float alpha[3][8];
  float o;
  {
    float sc[3][8];
    float mx = -3.0e38f;
#pragma unroll
    for (int jt = 0; jt < 3; ++jt)
#pragma unroll
      for (int r = 0; r < 8; ++r) {
        int j = 16 * jt + 8 * hi + r;
        float v = startT[j] + emis[((size_t)bb * SEQ) * NL + j];
        sc[jt][r] = v;
        mx = fmaxf(mx, v);
      }
    mx = fmaxf(mx, permx16_f32(mx));          // per-batch max across hi groups
#pragma unroll
    for (int jt = 0; jt < 3; ++jt)
#pragma unroll
      for (int r = 0; r < 8; ++r) alpha[jt][r] = __expf(sc[jt][r] - mx);
    o = mx;
  }

  // ---- prefetch emissions/mask for t = 1 (register double buffer) ----
  float cur[3][8];
#pragma unroll
  for (int jt = 0; jt < 3; ++jt)
#pragma unroll
    for (int r = 0; r < 8; ++r)
      cur[jt][r] = emis[((size_t)bb * SEQ + 1) * NL + 16 * jt + 8 * hi + r];
  int mcur = mask[bb * SEQ + 1];

  for (int t = 1; t < SEQ; ++t) {
    // Issue next step's loads early; independent of the recurrence chain.
    float nxt[3][8];
    int mnxt = 0;
    if (t + 1 < SEQ) {
#pragma unroll
      for (int jt = 0; jt < 3; ++jt)
#pragma unroll
        for (int r = 0; r < 8; ++r)
          nxt[jt][r] = emis[((size_t)bb * SEQ + (t + 1)) * NL + 16 * jt + 8 * hi + r];
      mnxt = mask[bb * SEQ + t + 1];
    } else {
#pragma unroll
      for (int jt = 0; jt < 3; ++jt)
#pragma unroll
        for (int r = 0; r < 8; ++r) nxt[jt][r] = 0.0f;
    }

    // ---- Build B (K x 16-batch, f16) from alpha, dword-granular ----
    // B VGPR g (halves 2g,2g+1) needs packed pair of tile (2kc+hi) regs
    // (2g,2g+1) from source group 0; VGPR g+4 the same pair from group 1.
    // kc=1 pairs tile 2 with an explicit ZERO tile (K = 48..63 padding).
    unsigned P0[4], P1[4], P2[4];
#pragma unroll
    for (int g = 0; g < 4; ++g) {
      P0[g] = pack2h(alpha[0][2 * g], alpha[0][2 * g + 1]);
      P1[g] = pack2h(alpha[1][2 * g], alpha[1][2 * g + 1]);
      P2[g] = pack2h(alpha[2][2 * g], alpha[2][2 * g + 1]);
    }
    v8u b0u, b1u;
#pragma unroll
    for (int g = 0; g < 4; ++g) {
      unsigned y0 = permx16_u32(P0[g]);
      unsigned y1 = permx16_u32(P1[g]);
      unsigned y2 = permx16_u32(P2[g]);
      b0u[g]     = hi ? y1    : P0[g];   // tile(0+hi) @ source group 0
      b0u[g + 4] = hi ? P1[g] : y0;      // tile(0+hi) @ source group 1
      b1u[g]     = hi ? 0u    : P2[g];   // tile(2+hi) @ group 0 (tile3 = 0)
      b1u[g + 4] = hi ? 0u    : y2;      // tile(2+hi) @ group 1 (tile3 = 0)
    }
    v16h Bm0 = __builtin_bit_cast(v16h, b0u);
    v16h Bm1 = __builtin_bit_cast(v16h, b1u);

    // ---- D = E^T @ alpha : 6x v_wmma_f32_16x16x32_f16 ----
    v8f c[3];
#pragma unroll
    for (int jt = 0; jt < 3; ++jt) { v8f z = {}; c[jt] = z; }
#pragma unroll
    for (int jt = 0; jt < 3; ++jt)
      c[jt] = __builtin_amdgcn_wmma_f32_16x16x32_f16(
          false, A[jt][0], false, Bm0, (short)0, c[jt], false, false);
#pragma unroll
    for (int jt = 0; jt < 3; ++jt)
      c[jt] = __builtin_amdgcn_wmma_f32_16x16x32_f16(
          false, A[jt][1], false, Bm1, (short)0, c[jt], false, false);

    // ---- w = D .* exp(emit_t); per-batch max; rescale; accumulate log ----
    float mx = 0.0f;
#pragma unroll
    for (int jt = 0; jt < 3; ++jt)
#pragma unroll
      for (int r = 0; r < 8; ++r) {
        float v = c[jt][r] * __expf(cur[jt][r]);
        c[jt][r] = v;
        mx = fmaxf(mx, v);
      }
    mx = fmaxf(mx, permx16_f32(mx));
    float rinv = __builtin_amdgcn_rcpf(mx);
    bool m = (mcur != 0);
#pragma unroll
    for (int jt = 0; jt < 3; ++jt)
#pragma unroll
      for (int r = 0; r < 8; ++r)
        alpha[jt][r] = m ? c[jt][r] * rinv : alpha[jt][r];
    o = m ? (o + __logf(mx)) : o;

    // rotate double buffer
#pragma unroll
    for (int jt = 0; jt < 3; ++jt)
#pragma unroll
      for (int r = 0; r < 8; ++r) cur[jt][r] = nxt[jt][r];
    mcur = mnxt;
  }

  // ---- finalize: fwd[b] = o + log( sum_j alpha[j,b] * exp(endT[j]) ) ----
  float s = 0.0f;
#pragma unroll
  for (int jt = 0; jt < 3; ++jt)
#pragma unroll
    for (int r = 0; r < 8; ++r) {
      int j = 16 * jt + 8 * hi + r;
      s += alpha[jt][r] * __expf(endT[j]);
    }
  s += permx16_f32(s);
  float f = o + __logf(s);
  if (hi == 0) fwd_out[bb] = f;
}

// ============================================================================
// Gold score. One wave per batch; lanes stride over time; deterministic
// butterfly reduction.
// ============================================================================
__global__ __launch_bounds__(32) void crf_gold_kernel(
    const float* __restrict__ emis, const int* __restrict__ labels,
    const int* __restrict__ mask, const float* __restrict__ trans,
    const float* __restrict__ startT, const float* __restrict__ endT,
    float* __restrict__ gold_out)
{
  const int b    = blockIdx.x;
  const int lane = threadIdx.x;
  const float* eb = emis + (size_t)b * SEQ * NL;
  const int* lb = labels + b * SEQ;
  const int* mb = mask + b * SEQ;

  float acc = 0.0f;
  int msum = 0;
  for (int t = lane; t < SEQ; t += 32) {
    int lt = lb[t];
    int mk = mb[t];
    msum += mk;
    if (t > 0) {
      int lp = lb[t - 1];
      acc += (eb[(size_t)t * NL + lt] + trans[lt * NL + lp]) * (float)mk;
    }
  }
#pragma unroll
  for (int d = 16; d >= 1; d >>= 1) {
    acc  += __shfl_xor(acc, d, 32);
    msum += __shfl_xor(msum, d, 32);
  }
  if (lane == 0) {
    int len = msum - 1;                     // lengths = sum(mask) - 1
    int l0 = lb[0];
    int ll = lb[len];
    gold_out[b] = startT[l0] + eb[l0] + acc + endT[ll];
  }
}

// ============================================================================
// Deterministic final mean over 512 batches: fixed-order LDS tree.
// ============================================================================
__global__ __launch_bounds__(256) void crf_reduce_kernel(
    const float* __restrict__ fwd, const float* __restrict__ gold,
    float* __restrict__ out)
{
  __shared__ float sm[256];
  int i = threadIdx.x;
  float v = (fwd[i] - gold[i]) + (fwd[i + 256] - gold[i + 256]);
  sm[i] = v;
  __syncthreads();
  for (int s = 128; s > 0; s >>= 1) {
    if (i < s) sm[i] += sm[i + s];
    __syncthreads();
  }
  if (i == 0) out[0] = sm[0] * (1.0f / 512.0f);
}

extern "C" void kernel_launch(void* const* d_in, const int* in_sizes, int n_in,
                              void* d_out, int out_size, void* d_ws, size_t ws_size,
                              hipStream_t stream)
{
  (void)in_sizes; (void)n_in; (void)out_size; (void)ws_size;
  const float* emis   = (const float*)d_in[0];
  const int*   labels = (const int*)d_in[1];
  const int*   mask   = (const int*)d_in[2];
  const float* trans  = (const float*)d_in[3];
  const float* startT = (const float*)d_in[4];
  const float* endT   = (const float*)d_in[5];

  float* fwd  = (float*)d_ws;          // [BATCH]
  float* gold = fwd + BATCH;           // [BATCH]

  crf_gold_kernel<<<BATCH, 32, 0, stream>>>(emis, labels, mask, trans, startT, endT, gold);
  crf_forward_kernel<<<BATCH / 16, 32, 0, stream>>>(emis, mask, trans, startT, endT, fwd);
  crf_reduce_kernel<<<1, 256, 0, stream>>>(fwd, gold, (float*)d_out);
}
